// LocationAwareAttention_24704651887073
// MI455X (gfx1250) — matrix operand order repaired and checked by reading, not compiled, not run on hardware
//
#include <hip/hip_runtime.h>
#include <hip/hip_bf16.h>
#include <math.h>

typedef __attribute__((ext_vector_type(16))) _Float16 v16h;
typedef __attribute__((ext_vector_type(8)))  _Float16 v8h;
typedef __attribute__((ext_vector_type(2)))  _Float16 v2h;
typedef __attribute__((ext_vector_type(8)))  float    v8f;

namespace {
constexpr int B_    = 32;
constexpr int T_    = 2000;
constexpr int ENC_  = 512;
constexpr int ATT_  = 512;
constexpr int CCH_  = 32;
constexpr int OUT_  = 512;
constexpr int KW_   = 100;     // conv half-width
constexpr int TAPS_ = 201;     // 2*KW_+1
constexpr int MROWS = B_ * T_; // 64000 (T_ % 16 == 0 -> 16-row tiles never cross batches)
constexpr int CHUNKS = 10;
constexpr int TCHUNK = T_ / CHUNKS; // 200
constexpr int APITCH = 40;          // LDS A-tile pitch in halves (80B, conflict-free frag reads)
constexpr int ABUF   = 32 * APITCH; // halves per buffer
}

// ---------------- prep: transpose + convert weights to f16 -------------------
__global__ void k_weights(const float* __restrict__ Wenc, const float* __restrict__ Wloc,
                          _Float16* __restrict__ WencT, _Float16* __restrict__ WlocT) {
  int idx = blockIdx.x * 256 + threadIdx.x;
  if (idx < ENC_ * ATT_) {
    int a = idx >> 9, k = idx & 511;          // WencT[a][k] = Wenc[k][a]
    WencT[(size_t)a * ENC_ + k] = (_Float16)Wenc[(size_t)k * ATT_ + a];
  }
  int idx2 = idx - ENC_ * ATT_;
  if (idx2 >= 0 && idx2 < CCH_ * ATT_) {
    int a = idx2 >> 5, c = idx2 & 31;         // WlocT[a][c] = Wloc[c][a]
    WlocT[(size_t)a * CCH_ + c] = (_Float16)Wloc[(size_t)c * ATT_ + a];
  }
}

// bias[b][a] = b_enc[a] + b_dec[a] + b_loc[a] + dec_states[b,:] @ W_dec[:,a]
__global__ void k_bias(const float* __restrict__ dec, const float* __restrict__ Wdec,
                       const float* __restrict__ benc, const float* __restrict__ bdec,
                       const float* __restrict__ bloc, float* __restrict__ bias) {
  int idx = blockIdx.x * 256 + threadIdx.x;   // 32*512
  int b = idx >> 9, a = idx & 511;
  float s = benc[a] + bdec[a] + bloc[a];
  const float* dr = dec + (size_t)b * ATT_;
  for (int k = 0; k < ATT_; ++k) s += dr[k] * Wdec[(size_t)k * ATT_ + a];
  bias[idx] = s;
}

// conv of prev_attn (piecewise-constant 1/L) == (1/L) * contiguous weight-range sum
__global__ void k_conv(const float* __restrict__ convw, const int* __restrict__ enc_len,
                       _Float16* __restrict__ convF) {
  int r = blockIdx.x * 256 + threadIdx.x;
  if (r >= MROWS) return;
  int b = r / T_, t = r - b * T_;
  int L = enc_len[b];
  float inv = 1.0f / (float)L;
  int i0 = KW_ - t;          if (i0 < 0) i0 = 0;
  int i1 = KW_ - t + L - 1;  if (i1 > TAPS_ - 1) i1 = TAPS_ - 1;
  _Float16* out = convF + (size_t)r * CCH_;
  for (int c = 0; c < CCH_; ++c) {
    float s = 0.0f;
    const float* w = convw + (size_t)c * TAPS_;
    for (int i = i0; i <= i1; ++i) s += w[i];
    out[c] = (_Float16)(s * inv);
  }
}

// ---------------- fused scores GEMM (WMMA f16, LDS-staged A) -----------------
// e[row] = w_attn . tanh( enc[row,:]@WencT^T + convF[row,:]@WlocT^T + bias[b,:] )
// Block: 512 thr = 16 waves; 32 rows (2 m-tiles) x 512 cols (8 n-groups of 64).
// Each wave: 1 m-tile x 4 WMMA col-tiles (acc = 32 VGPRs, stays pinned).
// Pipeline loop is branch-free (last iteration peeled) so the allocator does
// not need accumulator phi copies on the backedge.
__global__ __launch_bounds__(512)
void k_scores(const float* __restrict__ enc,
              const _Float16* __restrict__ WencT,
              const _Float16* __restrict__ WlocT,
              const _Float16* __restrict__ convF,
              const float* __restrict__ bias,
              const float* __restrict__ wattn,
              float* __restrict__ logits) {
  __shared__ __align__(16) _Float16 Atile[2 * ABUF];
  __shared__ float e_lds[32];
  if (threadIdx.x < 32) e_lds[threadIdx.x] = 0.0f;

  const int lane = threadIdx.x & 31;
  const int wid  = threadIdx.x >> 5;   // 0..15
  const int midx = wid & 1;            // which 16-row tile
  const int ng   = wid >> 1;           // which 64-col group (0..7)
  const int n    = lane & 15;          // A row-in-tile / B column-in-tile
  const int kh   = lane >> 4;          // K-half select
  const int ncol0 = ng * 64;

  const int rowBase = blockIdx.x * 32;
  const int rowW = rowBase + midx * 16;
  const int b    = rowW / T_;               // tile never crosses batch (T_%16==0)
  const int rowA = rowW + n;

  // staging map: thread -> (row, 2 contiguous cols) of the 32x32 tile
  const int srow  = threadIdx.x >> 4;        // 0..31
  const int scolh = (threadIdx.x & 15) * 2;  // 0..30
  const float* gsrc = enc + (size_t)(rowBase + srow) * ENC_ + scolh;
  _Float16* sdst = &Atile[srow * APITCH + scolh];

  // B fragment base: column (ncol0 + n), j-tile & K offsets folded into imm offsets
  const _Float16* bcol = WencT + (size_t)(ncol0 + n) * ENC_ + kh * 16;

  v8f acc[4];
#pragma unroll
  for (int j = 0; j < 4; ++j) {
    const float bb = bias[(size_t)b * ATT_ + ncol0 + j * 16 + n];
#pragma unroll
    for (int i = 0; i < 8; ++i) acc[j][i] = bb;
  }

  // stage first K-tile into buffer 0
  {
    float2 v = *reinterpret_cast<const float2*>(gsrc);
    v2h h; h[0] = (_Float16)v.x; h[1] = (_Float16)v.y;
    *reinterpret_cast<v2h*>(sdst) = h;
  }
  __syncthreads();

  const _Float16* afb = &Atile[(midx * 16 + n) * APITCH];

  // steady-state: iterations 0..14 prefetch tile s+1 unconditionally
  for (int s = 0; s < 15; ++s) {
    const int k0 = s * 32;
    const float2 nv = *reinterpret_cast<const float2*>(gsrc + k0 + 32);

    // A frag from LDS: lane holds K in [16kh,16kh+16)
    const int buf = (s & 1) * ABUF;
    const v8h c0 = *reinterpret_cast<const v8h*>(afb + buf + kh * 16);
    const v8h c1 = *reinterpret_cast<const v8h*>(afb + buf + kh * 16 + 8);
    v16h a;
#pragma unroll
    for (int e = 0; e < 8; ++e) { a[e] = c0[e]; a[8 + e] = c1[e]; }

#pragma unroll
    for (int j = 0; j < 4; ++j) {
      const v16h bf = *reinterpret_cast<const v16h*>(bcol + (size_t)j * 16 * ENC_ + k0);
      acc[j] = __builtin_amdgcn_wmma_f32_16x16x32_f16(
          false, a, false, bf, (short)0, acc[j], false, false);
    }

    v2h h; h[0] = (_Float16)nv.x; h[1] = (_Float16)nv.y;
    *reinterpret_cast<v2h*>(sdst + ((s + 1) & 1) * ABUF) = h;
    __syncthreads();
  }

  // tail: s = 15, compute only (buffer 1)
  {
    const int k0 = 15 * 32;
    const v8h c0 = *reinterpret_cast<const v8h*>(afb + ABUF + kh * 16);
    const v8h c1 = *reinterpret_cast<const v8h*>(afb + ABUF + kh * 16 + 8);
    v16h a;
#pragma unroll
    for (int e = 0; e < 8; ++e) { a[e] = c0[e]; a[8 + e] = c1[e]; }
#pragma unroll
    for (int j = 0; j < 4; ++j) {
      const v16h bf = *reinterpret_cast<const v16h*>(bcol + (size_t)j * 16 * ENC_ + k0);
      acc[j] = __builtin_amdgcn_wmma_f32_16x16x32_f16(
          false, a, false, bf, (short)0, acc[j], false, false);
    }
  }

  // conv-feature contribution: single K=32 step (tiny; direct global loads)
  {
    const _Float16* cf = convF + (size_t)rowA * CCH_;
    v16h a2;
#pragma unroll
    for (int e = 0; e < 8; ++e) { a2[e] = cf[kh * 16 + e]; a2[8 + e] = cf[kh * 16 + 8 + e]; }
#pragma unroll
    for (int j = 0; j < 4; ++j) {
      const v16h b2 = *reinterpret_cast<const v16h*>(
          WlocT + (size_t)(ncol0 + j * 16 + n) * CCH_ + kh * 16);
      acc[j] = __builtin_amdgcn_wmma_f32_16x16x32_f16(
          false, a2, false, b2, (short)0, acc[j], false, false);
    }
  }

  // epilogue: tanh, dot with w_attn, reduce columns
  // C layout: VGPR i -> row (i + 8*kh), col = ncol0 + j*16 + n
  float wv[4];
#pragma unroll
  for (int j = 0; j < 4; ++j) wv[j] = wattn[ncol0 + j * 16 + n];

#pragma unroll
  for (int i = 0; i < 8; ++i) {
    float part = 0.0f;
#pragma unroll
    for (int j = 0; j < 4; ++j) part += wv[j] * tanhf(acc[j][i]);
    part += __shfl_xor(part, 1, 32);
    part += __shfl_xor(part, 2, 32);
    part += __shfl_xor(part, 4, 32);
    part += __shfl_xor(part, 8, 32);
    if (n == 0) atomicAdd(&e_lds[midx * 16 + kh * 8 + i], part);
  }
  __syncthreads();
  if (threadIdx.x < 32)
    logits[(size_t)blockIdx.x * 32 + threadIdx.x] = e_lds[threadIdx.x];
}

// ---------------- softmax over T with length mask ----------------------------
__global__ __launch_bounds__(256)
void k_softmax(const float* __restrict__ logits, const int* __restrict__ enc_len,
               float* __restrict__ attn) {
  __shared__ float smax[8];
  __shared__ float ssum[8];
  const int b = blockIdx.x;
  const int L = enc_len[b];
  const float* lg = logits + (size_t)b * T_;
  const int lane = threadIdx.x & 31, wid = threadIdx.x >> 5;

  float m = -INFINITY;
  for (int t = threadIdx.x; t < T_; t += 256)
    if (t < L) m = fmaxf(m, lg[t]);
  for (int off = 16; off > 0; off >>= 1) m = fmaxf(m, __shfl_xor(m, off, 32));
  if (lane == 0) smax[wid] = m;
  __syncthreads();
  if (threadIdx.x == 0) {
    float mm = smax[0];
    for (int i = 1; i < 8; ++i) mm = fmaxf(mm, smax[i]);
    smax[0] = mm;
  }
  __syncthreads();
  m = smax[0];

  float s = 0.0f;
  for (int t = threadIdx.x; t < T_; t += 256)
    if (t < L) s += expf(lg[t] - m);
  for (int off = 16; off > 0; off >>= 1) s += __shfl_xor(s, off, 32);
  if (lane == 0) ssum[wid] = s;
  __syncthreads();
  if (threadIdx.x == 0) {
    float ss = 0.0f;
    for (int i = 0; i < 8; ++i) ss += ssum[i];
    ssum[0] = ss;
  }
  __syncthreads();
  const float rinv = 1.0f / ssum[0];

  float* at = attn + (size_t)b * T_;
  for (int t = threadIdx.x; t < T_; t += 256)
    at[t] = (t < L) ? expf(lg[t] - m) * rinv : 0.0f;
}

// ---------------- context = attn @ enc_states (chunked, deterministic) -------
__global__ __launch_bounds__(256)
void k_ctxpart(const float* __restrict__ enc, const float* __restrict__ attn,
               float* __restrict__ ctxp) {
  const int b = blockIdx.x, ch = blockIdx.y, e = threadIdx.x;
  const float* A = attn + (size_t)b * T_ + (size_t)ch * TCHUNK;
  const float* E = enc + ((size_t)b * T_ + (size_t)ch * TCHUNK) * ENC_;
  float a0 = 0.0f, a1 = 0.0f;
  for (int t = 0; t < TCHUNK; ++t) {
    const float w = A[t];
    const float* er = E + (size_t)t * ENC_;
    a0 += w * er[e];
    a1 += w * er[e + 256];
  }
  float* o = ctxp + ((size_t)ch * B_ + b) * ENC_;
  o[e] = a0;
  o[e + 256] = a1;
}

__global__ void k_ctxred(const float* __restrict__ ctxp, float* __restrict__ ctx) {
  int i = blockIdx.x * 256 + threadIdx.x; // 32*512
  float s = 0.0f;
  for (int c = 0; c < CHUNKS; ++c) s += ctxp[(size_t)c * B_ * ENC_ + i];
  ctx[i] = s;
}

// ---------------- out = ctx @ W_out + b_out ----------------------------------
__global__ void k_outgemm(const float* __restrict__ ctx, const float* __restrict__ Wout,
                          const float* __restrict__ bout, float* __restrict__ out) {
  int idx = blockIdx.x * 256 + threadIdx.x; // 32*512
  int b = idx >> 9, o = idx & 511;
  float s = bout[o];
  const float* cr = ctx + (size_t)b * ENC_;
  for (int e = 0; e < ENC_; ++e) s += cr[e] * Wout[(size_t)e * OUT_ + o];
  out[idx] = s;
}

extern "C" void kernel_launch(void* const* d_in, const int* in_sizes, int n_in,
                              void* d_out, int out_size, void* d_ws, size_t ws_size,
                              hipStream_t stream) {
  (void)in_sizes; (void)n_in; (void)out_size; (void)ws_size;
  const float* enc     = (const float*)d_in[0];
  const int*   enc_len = (const int*)  d_in[1];
  const float* dec     = (const float*)d_in[2];
  const float* Wenc    = (const float*)d_in[3];
  const float* benc    = (const float*)d_in[4];
  const float* Wdec    = (const float*)d_in[5];
  const float* bdec    = (const float*)d_in[6];
  const float* convw   = (const float*)d_in[7];
  const float* Wloc    = (const float*)d_in[8];
  const float* bloc    = (const float*)d_in[9];
  const float* wattn   = (const float*)d_in[10];
  const float* Wout    = (const float*)d_in[11];
  const float* bout    = (const float*)d_in[12];

  // workspace carve (all sizes multiples of 256B; hipMalloc base is 256B-aligned)
  uint8_t* p = (uint8_t*)d_ws;
  _Float16* WencT = (_Float16*)p; p += (size_t)ENC_ * ATT_ * 2;        // 512 KB
  _Float16* WlocT = (_Float16*)p; p += (size_t)CCH_ * ATT_ * 2;        // 32 KB
  float*    bias  = (float*)p;    p += (size_t)B_ * ATT_ * 4;          // 64 KB
  _Float16* convF = (_Float16*)p; p += (size_t)MROWS * CCH_ * 2;       // 4 MB
  float*    logit = (float*)p;    p += (size_t)MROWS * 4;              // 250 KB
  float*    ctxp  = (float*)p;    p += (size_t)CHUNKS * B_ * ENC_ * 4; // 640 KB
  float*    ctx   = (float*)p;    p += (size_t)B_ * ENC_ * 4;          // 64 KB

  float* out_ctx  = (float*)d_out;            // [B, OUT]
  float* out_attn = out_ctx + B_ * OUT_;      // [B, T]

  k_weights<<<(ENC_ * ATT_ + CCH_ * ATT_ + 255) / 256, 256, 0, stream>>>(Wenc, Wloc, WencT, WlocT);
  k_bias   <<<(B_ * ATT_) / 256, 256, 0, stream>>>(dec, Wdec, benc, bdec, bloc, bias);
  k_conv   <<<(MROWS + 255) / 256, 256, 0, stream>>>(convw, enc_len, convF);
  k_scores <<<MROWS / 32, 512, 0, stream>>>(enc, WencT, WlocT, convF, bias, wattn, logit);
  k_softmax<<<B_, 256, 0, stream>>>(logit, enc_len, out_attn);
  dim3 gctx(B_, CHUNKS);
  k_ctxpart<<<gctx, 256, 0, stream>>>(enc, out_attn, ctxp);
  k_ctxred <<<(B_ * ENC_) / 256, 256, 0, stream>>>(ctxp, ctx);
  k_outgemm<<<(B_ * OUT_) / 256, 256, 0, stream>>>(ctx, Wout, bout, out_ctx);
}
